// Encoder_90005334655635
// MI455X (gfx1250) — matrix operand order repaired and checked by reading, not compiled, not run on hardware
//
#include <hip/hip_runtime.h>

#define VOCAB 128
#define EMB   256
#define HID   256
#define BATCH 4096
#define TSTEP 30
#define ROWS  32            // batch rows per block
#define NBLK  (BATCH / ROWS)

typedef __attribute__((ext_vector_type(16))) __bf16 v16bf;
typedef __attribute__((ext_vector_type(8)))  __bf16 v8bf;
typedef __attribute__((ext_vector_type(8)))  float  v8f;

// global (addrspace 1) views so loads lower to global_load_*, not flat_load_*
typedef unsigned short __attribute__((address_space(1))) gus_t;
typedef v16bf          __attribute__((address_space(1))) gv16bf_t;

__device__ __forceinline__ unsigned short f2bf(float x) {
    unsigned u = __float_as_uint(x);
    unsigned r = u + 0x7FFFu + ((u >> 16) & 1u);   // round-to-nearest-even
    return (unsigned short)(r >> 16);
}
__device__ __forceinline__ float sigm(float x) {
    return 1.0f / (1.0f + __expf(-x));
}
__device__ __forceinline__ float tanh_fast(float x) {
    float ax = fabsf(x);
    float e  = __expf(-2.0f * ax);
    float r  = (1.0f - e) / (1.0f + e);
    return copysignf(r, x);
}

// ---------------------------------------------------------------------------
// Pack W_hh (768x256 f32, row-major) into bf16 WMMA B-operand lane layout.
// B tile (nt, kc): lane l holds N = nt*16 + (l&15), K = kc*32 + (l>>4)*16 + i,
// i = 0..15 -> contiguous 32B per lane, 1KB coalesced per wave load.
// ---------------------------------------------------------------------------
__global__ void pack_whh_kernel(const float* __restrict__ Whh,
                                unsigned short* __restrict__ Wp) {
    int id = blockIdx.x * blockDim.x + threadIdx.x;   // 48 nt * 8 kc * 32 lanes
    if (id >= 48 * 8 * 32) return;
    int lane = id & 31;
    int kc   = (id >> 5) & 7;
    int nt   = id >> 8;
    int N    = nt * 16 + (lane & 15);
    int koff = (lane >> 4) * 16;
    unsigned short* dst = Wp + id * 16;
    const float* src = Whh + N * HID + kc * 32 + koff;
#pragma unroll
    for (int i = 0; i < 16; ++i) dst[i] = f2bf(src[i]);
}

// ---------------------------------------------------------------------------
// G[v][c] = dot(emb[v,:], W_ih[c,:]) + b_ih[c]   (128 x 768, fp32)
// Folds the entire input GEMM of the GRU into a per-token lookup table.
// ---------------------------------------------------------------------------
__global__ void build_g_kernel(const float* __restrict__ emb,
                               const float* __restrict__ Wih,
                               const float* __restrict__ bih,
                               float* __restrict__ G) {
    int id = blockIdx.x * blockDim.x + threadIdx.x;   // 128*768
    if (id >= VOCAB * 3 * HID) return;
    int v = id / (3 * HID);
    int c = id % (3 * HID);
    const float* e = emb + v * EMB;
    const float* w = Wih + c * EMB;
    float s = bih[c];
#pragma unroll 4
    for (int k = 0; k < EMB; ++k) s = fmaf(e[k], w[k], s);
    G[id] = s;
}

// ---------------------------------------------------------------------------
// Persistent GRU: each block owns 32 batch rows for all 30 timesteps.
// Per step: gh = h @ W_hh^T via v_wmma_f32_16x16x32_bf16, then fused gates.
// B fragments re-read from L2 each step; opaque uniform base defeats LICM
// (no ~384-VGPR weight hoist/spill) while the addrspace(1) cast keeps the
// loads on the global path (LOADcnt only, saddr+imm addressing) instead of
// flat (which would couple them to DScnt and the LDS A-fragment waits).
// ---------------------------------------------------------------------------
__global__ __launch_bounds__(256)
void gru_main_kernel(const int* __restrict__ tok,
                     const float* __restrict__ h0,
                     const float* __restrict__ bhh,
                     const unsigned short* __restrict__ Wp,
                     const float* __restrict__ G,
                     float* __restrict__ out) {
    __shared__ float h_f32[ROWS][HID];                       // 32 KB fp32 state
    __shared__ __align__(16) unsigned short h_bf[ROWS][264]; // 16.5 KB bf16 copy (padded)
    __shared__ float s_bhh[3 * HID];                         // 3 KB
    __shared__ int   s_tok[ROWS];

    const int tid     = threadIdx.x;
    const int lane    = tid & 31;
    const int wave    = tid >> 5;            // 8 waves
    const int rowbase = blockIdx.x * ROWS;

    // --- init: biases + h0 (fp32 + bf16) ---
    for (int i = tid; i < 3 * HID; i += 256) s_bhh[i] = bhh[i];
    for (int i = tid; i < ROWS * HID; i += 256) {
        int r = i >> 8, c = i & (HID - 1);
        float v = h0[(size_t)(rowbase + r) * HID + c];
        h_f32[r][c] = v;
        h_bf[r][c]  = f2bf(v);
    }
    __syncthreads();

    const int aM    = lane & 15;             // A-operand row within 16-tile
    const int akoff = (lane >> 4) * 16;      // A-operand K sub-offset
    const int cN    = lane & 15;             // C-layout column
    const int cMo   = (lane >> 4) * 8;       // C-layout row offset

#pragma unroll 1
    for (int t = 0; t < TSTEP; ++t) {
        if (tid < ROWS) s_tok[tid] = tok[(size_t)(rowbase + tid) * TSTEP + t];

        // Opaque uniform base (defeats LICM of the t-invariant B loads),
        // cast back to an addrspace(1) pointer so loads stay global_load.
        unsigned long long wpu = (unsigned long long)Wp;
        asm volatile("" : "+s"(wpu));
        const gus_t* wbase = (const gus_t*)wpu + lane * 16;

        // ---- gh = h @ W_hh^T for this block's 32 rows ----
        // acc[jj][gate][mt] : 12 independent 16x16 f32 accumulator chains
        v8f acc[2][3][2];
        const v8f vz = {0.f, 0.f, 0.f, 0.f, 0.f, 0.f, 0.f, 0.f};
#pragma unroll
        for (int jj = 0; jj < 2; ++jj)
#pragma unroll
            for (int g = 0; g < 3; ++g)
#pragma unroll
                for (int mt = 0; mt < 2; ++mt) acc[jj][g][mt] = vz;

#pragma unroll
        for (int kc = 0; kc < 8; ++kc) {
            // A fragments: lanes 0-15 rows M, K k0..k0+15;
            // lanes 16-31 same rows, K k0+16..k0+31 (matches packed B layout).
            v16bf afr[2];
#pragma unroll
            for (int mt = 0; mt < 2; ++mt) {
                const unsigned short* p = &h_bf[mt * 16 + aM][kc * 32 + akoff];
                v8bf lo = *(const v8bf*)(p);
                v8bf hi = *(const v8bf*)(p + 8);
                afr[mt] = __builtin_shufflevector(lo, hi,
                            0, 1, 2, 3, 4, 5, 6, 7, 8, 9, 10, 11, 12, 13, 14, 15);
            }
#pragma unroll
            for (int jj = 0; jj < 2; ++jj) {
                const int jt = wave * 2 + jj;            // j-tile 0..15
#pragma unroll
                for (int g = 0; g < 3; ++g) {
                    const int nt = g * 16 + jt;          // gate-column tile 0..47
                    // element offset ((nt*8+kc)*32 + lane)*16 = lane*16 + const
                    v16bf bfr = *(const gv16bf_t*)(wbase + (nt * 8 + kc) * 512);
#pragma unroll
                    for (int mt = 0; mt < 2; ++mt) {
                        acc[jj][g][mt] = __builtin_amdgcn_wmma_f32_16x16x32_bf16(
                            false, afr[mt], false, bfr, (short)0, acc[jj][g][mt],
                            false, false);
                    }
                }
            }
        }

        __syncthreads();   // all h_bf reads done; s_tok visible

        // ---- fused gates + state update + output store ----
#pragma unroll
        for (int jj = 0; jj < 2; ++jj) {
            const int j = (wave * 2 + jj) * 16 + cN;     // hidden col 0..255
            const float bh_r = s_bhh[j];
            const float bh_z = s_bhh[HID + j];
            const float bh_n = s_bhh[2 * HID + j];
#pragma unroll
            for (int mt = 0; mt < 2; ++mt) {
#pragma unroll
                for (int e = 0; e < 8; ++e) {
                    const int row = mt * 16 + e + cMo;   // local batch row
                    const int tk  = s_tok[row];
                    const float* Gp = G + tk * (3 * HID) + j;
                    float gi_r = Gp[0];
                    float gi_z = Gp[HID];
                    float gi_n = Gp[2 * HID];
                    float gh_r = acc[jj][0][mt][e] + bh_r;
                    float gh_z = acc[jj][1][mt][e] + bh_z;
                    float gh_n = acc[jj][2][mt][e] + bh_n;
                    float rg = sigm(gi_r + gh_r);
                    float zg = sigm(gi_z + gh_z);
                    float ng = tanh_fast(fmaf(rg, gh_n, gi_n));
                    float ho = h_f32[row][j];
                    float hv = fmaf(zg, ho - ng, ng);    // (1-z)*n + z*h
                    h_f32[row][j] = hv;
                    h_bf[row][j]  = f2bf(hv);
                    const size_t gofs = (size_t)(rowbase + row) * HID + j;
                    // write-once streaming output: keep it out of hot L2 sets
                    __builtin_nontemporal_store(hv,
                        out + (size_t)t * BATCH * HID + gofs);
                    if (t == TSTEP - 1)
                        __builtin_nontemporal_store(hv,
                            out + (size_t)TSTEP * BATCH * HID + gofs);
                }
            }
        }
        __syncthreads();   // state writes done before next step's reads
    }
}

// ---------------------------------------------------------------------------
extern "C" void kernel_launch(void* const* d_in, const int* in_sizes, int n_in,
                              void* d_out, int out_size, void* d_ws, size_t ws_size,
                              hipStream_t stream) {
    const int*   tok = (const int*)d_in[0];     // input_t (4096,30)
    const float* h0  = (const float*)d_in[1];   // (1,4096,256)
    const float* emb = (const float*)d_in[2];   // (128,256)
    const float* Wih = (const float*)d_in[3];   // (768,256)
    const float* Whh = (const float*)d_in[4];   // (768,256)
    const float* bih = (const float*)d_in[5];   // (768,)
    const float* bhh = (const float*)d_in[6];   // (768,)
    float* out = (float*)d_out;

    float*          G  = (float*)d_ws;                                  // 128*768 f32
    unsigned short* Wp = (unsigned short*)((char*)d_ws + VOCAB * 3 * HID * sizeof(float));

    pack_whh_kernel<<<48, 256, 0, stream>>>(Whh, Wp);
    build_g_kernel<<<(VOCAB * 3 * HID + 255) / 256, 256, 0, stream>>>(emb, Wih, bih, G);
    gru_main_kernel<<<NBLK, 256, 0, stream>>>(tok, h0, bhh, Wp, G, out);
}
